// MDPBMP_metapath_specific_32298154066241
// MI455X (gfx1250) — matrix operand order repaired and checked by reading, not compiled, not run on hardware
//
#include <hip/hip_runtime.h>

typedef _Float16 v16h __attribute__((ext_vector_type(16)));
typedef float    v8f  __attribute__((ext_vector_type(8)));

constexpr int N_NODES = 50000;
constexpr int E  = 200000;
constexpr int L  = 4;
constexpr int H  = 8;
constexpr int D  = 64;
constexpr int HD = H * D;      // 512
constexpr float SLOPE = 0.01f;

// ---------------------------------------------------------------------------
// CDNA5 async gather: per-lane 16B copy global -> LDS, tracked by ASYNCcnt.
// dst-first syntax: global_load_async_to_lds_b128 vdst(LDS addr), v[addr64], off
// ---------------------------------------------------------------------------
__device__ __forceinline__ void async_ld16(unsigned lds_off, const float* g) {
    unsigned long long ga = (unsigned long long)(uintptr_t)g;
    asm volatile("global_load_async_to_lds_b128 %0, %1, off"
                 :: "v"(lds_off), "v"(ga) : "memory");
}
__device__ __forceinline__ void wait_async0() {
    asm volatile("s_wait_asynccnt 0x0" ::: "memory");
}

// ---------------------------------------------------------------------------
// 1) fp32 -> f16 weight conversion (W is [HD, D] row-major, k contiguous)
// ---------------------------------------------------------------------------
__global__ void k_convert_w(const float* __restrict__ W, _Float16* __restrict__ Wh) {
    int i = blockIdx.x * 256 + threadIdx.x;
    if (i < HD * D) Wh[i] = (_Float16)W[i];
}

// ---------------------------------------------------------------------------
// 2) init: out = 0, m = -inf, s = 0
// ---------------------------------------------------------------------------
__global__ void k_init(float* __restrict__ out, float* __restrict__ m, float* __restrict__ s) {
    int i = blockIdx.x * 256 + threadIdx.x;
    if (i < N_NODES * HD) out[i] = 0.0f;
    if (i < N_NODES * H) { m[i] = -__builtin_inff(); s[i] = 0.0f; }
}

// ---------------------------------------------------------------------------
// 3) GEMM kernel, two modes sharing the WMMA pipeline.
//    Block = 32 threads = 1 wave; wave owns 32 edges as two M=16 tiles.
//    Each B fragment feeds two independent WMMA chains (fills the f16-WMMA
//    hazard/co-exec slots that showed up as v_nop with a single tile) and
//    halves W traffic per output element.
//    MODE 0: fused attention logits (shfl_xor tree) -> alog[E,H]
//    MODE 1: out[dst[e], n] += vmax * alpha[e,h]   (f32 atomics, L2-resident)
// ---------------------------------------------------------------------------
template <int MODE>
__global__ __launch_bounds__(32) void k_gemm(
    const float* __restrict__ feat, const int* __restrict__ idx,
    const _Float16* __restrict__ Wh, const float* __restrict__ bias,
    const float* __restrict__ attn, float* __restrict__ alog,
    const float* __restrict__ alpha, const int* __restrict__ dst,
    float* __restrict__ out)
{
    __shared__ float Alds[32 * L * D];   // 32 edges * 1KB = 32KB
    const int lane = threadIdx.x & 31;
    const int e0   = blockIdx.x * 32;    // E = 6250 * 32, exact

    float* stage = &Alds[0];

    // ---- async gather: 32 edges * 4 pos * 16 float4 = 2048 x 16B copies ----
    for (int i = lane; i < 32 * L * (D / 4); i += 32) {
        int m   = i >> 6;          // 64 float4 per edge
        int rem = i & 63;
        int l   = rem >> 4;
        int d4  = rem & 15;
        int row = idx[(e0 + m) * L + l];
        const float* gp = feat + (size_t)row * D + d4 * 4;
        // low 32 bits of the flat address of an LDS location = LDS byte offset
        unsigned loff = (unsigned)(uintptr_t)(stage + (m * L + l) * D + d4 * 4);
        async_ld16(loff, gp);
    }
    wait_async0();
    __syncthreads();   // single-wave WG: S_NOP, ordering comes from asynccnt

    // ---- build f16 A fragments in contiguous k order (k = 16*hi + i), same as B ----
    // lane: row m = lane%16, half-group hi = lane/16
    // half i<8  -> k = kc*32 + i      + hi*16
    // half i>=8 -> k = kc*32 + i      + hi*16
    const int mrow = lane & 15;
    const int hi   = lane >> 4;
    v16h afrag[2][L][2];
#pragma unroll
    for (int mt = 0; mt < 2; ++mt) {
#pragma unroll
        for (int l = 0; l < L; ++l) {
#pragma unroll
            for (int c = 0; c < 2; ++c) {
                const float* ap = stage + ((mt * 16 + mrow) * L + l) * D + c * 32 + hi * 16;
                v16h a;
#pragma unroll
                for (int t = 0; t < 8; ++t) {
                    a[t]     = (_Float16)ap[t];
                    a[8 + t] = (_Float16)ap[8 + t];
                }
                afrag[mt][l][c] = a;
            }
        }
    }

    v8f apart[2];
#pragma unroll
    for (int mt = 0; mt < 2; ++mt)
#pragma unroll
        for (int v = 0; v < 8; ++v) apart[mt][v] = 0.0f;

    int   dn[2][8];
    float alpv[2][8];
    if (MODE == 1) {
#pragma unroll
        for (int mt = 0; mt < 2; ++mt)
#pragma unroll
            for (int v = 0; v < 8; ++v)
                dn[mt][v] = dst[e0 + mt * 16 + v + hi * 8];
    }

    // ---- loop over 32 N-tiles of 16 columns ----
    for (int nt = 0; nt < HD / 16; ++nt) {
        const int n = nt * 16 + mrow;        // this lane's output column

        if (MODE == 1 && (nt & 3) == 0) {
            const int h = nt >> 2;
#pragma unroll
            for (int mt = 0; mt < 2; ++mt)
#pragma unroll
                for (int v = 0; v < 8; ++v)
                    alpv[mt][v] = alpha[(e0 + mt * 16 + v + hi * 8) * H + h];
        }

        v8f vmax[2];
#pragma unroll
        for (int mt = 0; mt < 2; ++mt)
#pragma unroll
            for (int v = 0; v < 8; ++v) vmax[mt][v] = -__builtin_inff();

#pragma unroll
        for (int l = 0; l < L; ++l) {
            v8f acc0 = {};
            v8f acc1 = {};
#pragma unroll
            for (int c = 0; c < 2; ++c) {
                // B fragment: 16 contiguous halves of W row n;
                // lanes 0-15 cover K=kc*32..+15, lanes 16-31 K=kc*32+16..+31
                const _Float16* bp = Wh + n * D + c * 32 + hi * 16;
                v16h bfrag = *reinterpret_cast<const v16h*>(bp);
                acc0 = __builtin_amdgcn_wmma_f32_16x16x32_f16(
                    false, afrag[0][l][c], false, bfrag, (short)0, acc0, false, false);
                acc1 = __builtin_amdgcn_wmma_f32_16x16x32_f16(
                    false, afrag[1][l][c], false, bfrag, (short)0, acc1, false, false);
            }
#pragma unroll
            for (int v = 0; v < 8; ++v) {
                vmax[0][v] = fmaxf(vmax[0][v], acc0[v]);
                vmax[1][v] = fmaxf(vmax[1][v], acc1[v]);
            }
        }

        const float bv = bias[n];
#pragma unroll
        for (int mt = 0; mt < 2; ++mt)
#pragma unroll
            for (int v = 0; v < 8; ++v) vmax[mt][v] += bv;   // bias is l-invariant

        if (MODE == 0) {
            const float av = attn[n];
#pragma unroll
            for (int mt = 0; mt < 2; ++mt)
#pragma unroll
                for (int v = 0; v < 8; ++v) apart[mt][v] += vmax[mt][v] * av;

            if ((nt & 3) == 3) {              // head boundary: reduce 16 lanes
                const int h = nt >> 2;
#pragma unroll
                for (int mt = 0; mt < 2; ++mt) {
#pragma unroll
                    for (int v = 0; v < 8; ++v) {
                        float t = apart[mt][v];
                        t += __shfl_xor(t, 1, 32);
                        t += __shfl_xor(t, 2, 32);
                        t += __shfl_xor(t, 4, 32);
                        t += __shfl_xor(t, 8, 32);
                        if (mrow == 0) {
                            float lr = t > 0.0f ? t : SLOPE * t;
                            alog[(e0 + mt * 16 + v + hi * 8) * H + h] = lr;
                        }
                        apart[mt][v] = 0.0f;
                    }
                }
            }
        } else {
            // weighted scatter: C layout -> lane holds col n, rows m = v + hi*8
#pragma unroll
            for (int mt = 0; mt < 2; ++mt)
#pragma unroll
                for (int v = 0; v < 8; ++v)
                    atomicAdd(&out[(size_t)dn[mt][v] * HD + n],
                              vmax[mt][v] * alpv[mt][v]);
        }
    }
}

// ---------------------------------------------------------------------------
// 4) segment max via bit-trick float atomic max
// ---------------------------------------------------------------------------
__device__ inline void atomicMaxF(float* a, float v) {
    if (v >= 0.0f) atomicMax(reinterpret_cast<int*>(a), __float_as_int(v));
    else           atomicMin(reinterpret_cast<unsigned int*>(a),
                             (unsigned int)__float_as_int(v));
}

__global__ void k_segmax(const float* __restrict__ alog, const int* __restrict__ dst,
                         float* __restrict__ m) {
    int i = blockIdx.x * 256 + threadIdx.x;
    if (i >= E * H) return;
    int e = i >> 3, h = i & 7;
    atomicMaxF(&m[dst[e] * H + h], alog[i]);
}

// ---------------------------------------------------------------------------
// 5) in-place: a -> ae = exp(a - m[dst]); s[dst] += ae
// ---------------------------------------------------------------------------
__global__ void k_expsum(float* __restrict__ alog, const int* __restrict__ dst,
                         const float* __restrict__ m, float* __restrict__ s) {
    int i = blockIdx.x * 256 + threadIdx.x;
    if (i >= E * H) return;
    int e = i >> 3, h = i & 7;
    int dn = dst[e];
    float v = __expf(alog[i] - m[dn * H + h]);
    alog[i] = v;
    atomicAdd(&s[dn * H + h], v);
}

// ---------------------------------------------------------------------------
// 6) in-place: ae -> alpha = ae / s[dst]
// ---------------------------------------------------------------------------
__global__ void k_alpha(float* __restrict__ alog, const int* __restrict__ dst,
                        const float* __restrict__ s) {
    int i = blockIdx.x * 256 + threadIdx.x;
    if (i >= E * H) return;
    int e = i >> 3, h = i & 7;
    alog[i] = alog[i] / s[dst[e] * H + h];
}

// ---------------------------------------------------------------------------
extern "C" void kernel_launch(void* const* d_in, const int* in_sizes, int n_in,
                              void* d_out, int out_size, void* d_ws, size_t ws_size,
                              hipStream_t stream) {
    const float* feat = (const float*)d_in[0];
    const int*   idx  = (const int*)d_in[1];
    const int*   dst  = (const int*)d_in[2];
    const float* W    = (const float*)d_in[3];
    const float* b    = (const float*)d_in[4];
    const float* attn = (const float*)d_in[5];
    float* out = (float*)d_out;

    char* ws = (char*)d_ws;
    auto align256 = [](size_t x) { return (x + 255) & ~(size_t)255; };
    size_t off = 0;
    _Float16* Wh = (_Float16*)(ws + off); off = align256(off + (size_t)HD * D * sizeof(_Float16));
    float* alog  = (float*)(ws + off);    off = align256(off + (size_t)E * H * sizeof(float));
    float* m     = (float*)(ws + off);    off = align256(off + (size_t)N_NODES * H * sizeof(float));
    float* s     = (float*)(ws + off);    off = align256(off + (size_t)N_NODES * H * sizeof(float));

    k_convert_w<<<(HD * D + 255) / 256, 256, 0, stream>>>(W, Wh);
    k_init<<<(N_NODES * HD + 255) / 256, 256, 0, stream>>>(out, m, s);

    // phase 1: GEMM -> attention logits only (no eft materialization)
    k_gemm<0><<<E / 32, 32, 0, stream>>>(feat, idx, Wh, b, attn, alog,
                                         nullptr, nullptr, nullptr);
    k_segmax<<<(E * H + 255) / 256, 256, 0, stream>>>(alog, dst, m);
    k_expsum<<<(E * H + 255) / 256, 256, 0, stream>>>(alog, dst, m, s);
    k_alpha<<<(E * H + 255) / 256, 256, 0, stream>>>(alog, dst, s);

    // phase 2: recompute GEMM (feat/W are L2-resident), weighted atomic scatter
    k_gemm<1><<<E / 32, 32, 0, stream>>>(feat, idx, Wh, b, nullptr, nullptr,
                                         alog, dst, out);
}
